// PointDecoderSimple_14070312861812
// MI455X (gfx1250) — compile-verified, matrix operand order checked by reading, not run on hardware
//
#include <hip/hip_runtime.h>

typedef __attribute__((ext_vector_type(16))) _Float16 v16h;
typedef __attribute__((ext_vector_type(8)))  _Float16 v8h;
typedef __attribute__((ext_vector_type(8)))  float    v8f;

#define BS   8
#define SEQ  1024
#define DIM  256
#define NH   8
#define DH   32
#define NL   4
#define FFD  1024
#define PARTS 4
#define FULLS 1028
#define ROWS (BS*SEQ)   // 8192

__device__ __forceinline__ v8f wmma16(v16h a, v16h b, v8f c) {
    return __builtin_amdgcn_wmma_f32_16x16x32_f16(false, a, false, b, (short)0, c, false, false);
}

// four b128 f32 loads -> one f16 A/B fragment (k = base..base+7, base+16..base+23)
__device__ __forceinline__ v16h cvt16(float4 a0, float4 a1, float4 a2, float4 a3) {
    v16h r;
    r[0]=(_Float16)a0.x; r[1]=(_Float16)a0.y; r[2]=(_Float16)a0.z; r[3]=(_Float16)a0.w;
    r[4]=(_Float16)a1.x; r[5]=(_Float16)a1.y; r[6]=(_Float16)a1.z; r[7]=(_Float16)a1.w;
    r[8]=(_Float16)a2.x; r[9]=(_Float16)a2.y; r[10]=(_Float16)a2.z; r[11]=(_Float16)a2.w;
    r[12]=(_Float16)a3.x; r[13]=(_Float16)a3.y; r[14]=(_Float16)a3.z; r[15]=(_Float16)a3.w;
    return r;
}

__device__ __forceinline__ v16h load_frag_f32(const float* p) {
    return cvt16(*(const float4*)(p), *(const float4*)(p + 4),
                 *(const float4*)(p + 16), *(const float4*)(p + 20));
}

// ---------------- weight pre-pack: f32 [K,N] row-major -> WMMA B fragments ----------------
// layout: frag = tn*(K/32)+kk ; out[((frag*32 + lane)*16) + e] = W[k(lane,e)][tn*16 + lane&15]
__global__ void pack_w_kernel(const float* __restrict__ W, _Float16* __restrict__ out,
                              int K, int N) {
    int idx = blockIdx.x * 256 + threadIdx.x;
    if (idx >= K * N) return;
    int e    = idx & 15;
    int lane = (idx >> 4) & 31;
    int rest = idx >> 9;
    int kblocks = K >> 5;
    int tn = rest / kblocks, kk = rest % kblocks;
    int half = lane >> 4, l16 = lane & 15, base = half * 8;
    int i = e >> 1, j = e & 1;
    int k = kk * 32 + ((i >> 2) << 4) + base + 2 * (i & 3) + j;
    out[idx] = (_Float16)W[(size_t)k * N + tn * 16 + l16];
}

// ---------------- V transpose to f16: VT[b][d][s] = V[b,s,d] ----------------
__global__ void vtrans_kernel(const float* __restrict__ V, _Float16* __restrict__ VT) {
    int r = blockIdx.x;          // b*SEQ + s
    int d = threadIdx.x;         // 0..255
    int b = r >> 10, s = r & 1023;
    VT[((size_t)b * DIM + d) * SEQ + s] = (_Float16)V[(size_t)r * DIM + d];
}

// ---------------- pack: labels / position ----------------
__global__ void pack_kernel(const float* __restrict__ ratio,
                            int* __restrict__ labels, int* __restrict__ position,
                            int* __restrict__ lab_out) {
    int b = blockIdx.x;
    int cum[PARTS], prev[PARTS];
    int run = 0;
    for (int p = 0; p < PARTS; ++p) {
        prev[p] = (p == 0) ? 0 : cum[p - 1];
        run += (int)ceilf(ratio[b * PARTS + p] * (float)SEQ);
        cum[p] = run < SEQ ? run : SEQ;
    }
    for (int t = threadIdx.x; t < SEQ; t += blockDim.x) {
        int raw = 0;
        for (int p = 0; p < PARTS; ++p) raw += (t >= cum[p]) ? 1 : 0;
        bool valid = raw < PARTS;
        int lab = valid ? raw : 0;
        int pos = valid ? (t - prev[lab]) : 0;
        labels[b * SEQ + t]   = lab;
        position[b * SEQ + t] = pos;
        lab_out[b * SEQ + t]  = lab;
    }
}

// ---------------- embedding build ----------------
__global__ void embed_kernel(const float* __restrict__ pos_table,
                             const float* __restrict__ label_table,
                             const float* __restrict__ enc,
                             const int* __restrict__ labels,
                             const int* __restrict__ position,
                             float* __restrict__ x) {
    int r = blockIdx.x, d = threadIdx.x;
    int b = r >> 10;
    int lab = labels[r], pos = position[r];
    x[(size_t)r * DIM + d] = pos_table[(size_t)pos * DIM + d]
                           + enc[((size_t)b * PARTS + lab) * DIM + d]
                           + label_table[(size_t)lab * DIM + d];
}

// ---------------- memory slice copy ----------------
__global__ void copymem_kernel(const float* __restrict__ full, float* __restrict__ memc) {
    int r = blockIdx.x, d = threadIdx.x;
    int b = r >> 10, s = r & 1023;
    memc[(size_t)r * DIM + d] = full[((size_t)b * FULLS + PARTS + s) * DIM + d];
}

// ---------------- LayerNorm (optional ReLU) ----------------
__global__ void ln_kernel(const float* __restrict__ X, float* __restrict__ Y, int N, int relu) {
    __shared__ float s1[256], s2[256];
    int row = blockIdx.x;
    const float* xr = X + (size_t)row * N;
    float sum = 0.f, sq = 0.f;
    for (int i = threadIdx.x; i < N; i += 256) { float v = xr[i]; sum += v; sq += v * v; }
    s1[threadIdx.x] = sum; s2[threadIdx.x] = sq;
    __syncthreads();
    for (int o = 128; o > 0; o >>= 1) {
        if (threadIdx.x < o) { s1[threadIdx.x] += s1[threadIdx.x + o]; s2[threadIdx.x] += s2[threadIdx.x + o]; }
        __syncthreads();
    }
    float mean = s1[0] / (float)N;
    float var  = s2[0] / (float)N - mean * mean;
    float inv  = rsqrtf(var + 1e-5f);
    for (int i = threadIdx.x; i < N; i += 256) {
        float v = (xr[i] - mean) * inv;
        if (relu) v = fmaxf(v, 0.f);
        Y[(size_t)row * N + i] = v;
    }
}

// ---------------- WMMA GEMM: C = act(A @ W + bias) + res ----------------
// A [M,K] f32 row-major; Wp pre-packed f16 fragments. NT 16x16 tiles per wave (A reuse).
template<int NT, bool BIAS, bool RELU, bool RES>
__global__ void gemm_kernel(const float* __restrict__ A, const _Float16* __restrict__ Wp,
                            const float* __restrict__ bias, const float* __restrict__ res,
                            float* __restrict__ C, int M, int K, int N) {
    int wave = threadIdx.x >> 5, lane = threadIdx.x & 31;
    int tilesN = N >> 4;
    int ngroups = tilesN / NT;
    int g = blockIdx.x * 8 + wave;
    int tm = g / ngroups, tn0 = (g % ngroups) * NT;
    if (tm * 16 >= M) return;
    int half = lane >> 4, l16 = lane & 15, base = half * 8;
    int row = tm * 16 + l16;
    int kblocks = K >> 5;
    const float* ar = A + (size_t)row * K + base;
    v8f acc[NT];
    #pragma unroll
    for (int j = 0; j < NT; ++j) acc[j] = (v8f){};
    for (int kk = 0; kk < kblocks; ++kk) {
        v16h a = load_frag_f32(ar + kk * 32);
        #pragma unroll
        for (int j = 0; j < NT; ++j) {
            const _Float16* wp = Wp + ((((size_t)(tn0 + j) * kblocks + kk) * 32 + lane) << 4);
            v16h b = *(const v16h*)wp;
            acc[j] = wmma16(a, b, acc[j]);
        }
    }
    #pragma unroll
    for (int j = 0; j < NT; ++j) {
        int col = (tn0 + j) * 16 + l16;
        float bv = BIAS ? bias[col] : 0.f;
        #pragma unroll
        for (int r = 0; r < 8; ++r) {
            int m = tm * 16 + r + half * 8;
            float v = acc[j][r] + bv;
            if (RELU) v = fmaxf(v, 0.f);
            if (RES)  v += res[(size_t)m * N + col];
            C[(size_t)m * N + col] = v;
        }
    }
}

// ---------------- attention scores (one batch): SC[h,q,k] = (q.k)*scale ----------------
// Q-fragment reused across 4 key tiles.
__global__ void score_kernel(const float* __restrict__ Q, const float* __restrict__ Kb,
                             float* __restrict__ SC, int b, float scale) {
    int wave = threadIdx.x >> 5, lane = threadIdx.x & 31;
    int h = blockIdx.y;
    int tile = blockIdx.x * 8 + wave;        // 1024 = 64 q-tiles x 16 k-groups
    int tq = tile >> 4, tk0 = (tile & 15) * 4;
    int half = lane >> 4, l16 = lane & 15, base = half * 8;
    int qi = tq * 16 + l16;
    v16h a = load_frag_f32(Q + ((size_t)b * SEQ + qi) * DIM + h * DH + base);
    #pragma unroll
    for (int j = 0; j < 4; ++j) {
        int ki = (tk0 + j) * 16 + l16;
        v16h bb = load_frag_f32(Kb + ((size_t)b * SEQ + ki) * DIM + h * DH + base);
        v8f acc = {};
        acc = wmma16(a, bb, acc);
        #pragma unroll
        for (int r = 0; r < 8; ++r) {
            int m = tq * 16 + r + half * 8;
            SC[((size_t)h * SEQ + m) * SEQ + (tk0 + j) * 16 + l16] = acc[r] * scale;
        }
    }
}

// ---------------- masked softmax over rows of SC (one batch) ----------------
__global__ void softmax_kernel(float* __restrict__ SC, const int* __restrict__ labs,
                               const unsigned char* __restrict__ msk, int b) {
    __shared__ float red[256];
    int h = blockIdx.x >> 10, q = blockIdx.x & 1023;
    float* row = SC + ((size_t)(h * SEQ + q)) * SEQ;
    int lq = labs ? labs[b * SEQ + q] : 0;
    float vals[4];
    float mx = -3.4e38f;
    #pragma unroll
    for (int j = 0; j < 4; ++j) {
        int k = threadIdx.x + j * 256;
        float v = row[k];
        bool ok = labs ? (labs[b * SEQ + k] == lq) : (msk[(size_t)b * FULLS + PARTS + k] != 0);
        if (!ok) v = -1e9f;
        vals[j] = v;
        mx = fmaxf(mx, v);
    }
    red[threadIdx.x] = mx; __syncthreads();
    for (int o = 128; o > 0; o >>= 1) {
        if (threadIdx.x < o) red[threadIdx.x] = fmaxf(red[threadIdx.x], red[threadIdx.x + o]);
        __syncthreads();
    }
    mx = red[0]; __syncthreads();
    float s = 0.f;
    #pragma unroll
    for (int j = 0; j < 4; ++j) { vals[j] = __expf(vals[j] - mx); s += vals[j]; }
    red[threadIdx.x] = s; __syncthreads();
    for (int o = 128; o > 0; o >>= 1) {
        if (threadIdx.x < o) red[threadIdx.x] += red[threadIdx.x + o];
        __syncthreads();
    }
    float inv = 1.f / red[0];
    #pragma unroll
    for (int j = 0; j < 4; ++j) row[threadIdx.x + j * 256] = vals[j] * inv;
}

// ---------------- AO[b,q,h*32+n] = sum_k P[h,q,k] * V[b,k,h*32+n] ----------------
// P-fragment reused across both dh tiles of a head; V pre-transposed f16 [B,D,S].
__global__ void av_kernel(const float* __restrict__ P, const _Float16* __restrict__ VT,
                          float* __restrict__ AO, int b) {
    int wave = threadIdx.x >> 5, lane = threadIdx.x & 31;
    int tile = blockIdx.x * 8 + wave;        // 512 = 64 q-tiles x 8 heads
    int tq = tile >> 3, h = tile & 7;
    int half = lane >> 4, l16 = lane & 15, base = half * 8;
    int qi = tq * 16 + l16;
    const float* prow = P + ((size_t)h * SEQ + qi) * SEQ + base;
    const _Float16* vp0 = VT + ((size_t)b * DIM + h * DH + l16) * SEQ + base;
    const _Float16* vp1 = vp0 + (size_t)16 * SEQ;
    v8f acc0 = {}, acc1 = {};
    for (int kk = 0; kk < SEQ; kk += 32) {
        v16h a = load_frag_f32(prow + kk);
        v8h b0lo = *(const v8h*)(vp0 + kk), b0hi = *(const v8h*)(vp0 + kk + 16);
        v8h b1lo = *(const v8h*)(vp1 + kk), b1hi = *(const v8h*)(vp1 + kk + 16);
        v16h b0 = __builtin_shufflevector(b0lo, b0hi, 0,1,2,3,4,5,6,7,8,9,10,11,12,13,14,15);
        v16h b1 = __builtin_shufflevector(b1lo, b1hi, 0,1,2,3,4,5,6,7,8,9,10,11,12,13,14,15);
        acc0 = wmma16(a, b0, acc0);
        acc1 = wmma16(a, b1, acc1);
    }
    #pragma unroll
    for (int r = 0; r < 8; ++r) {
        int m = tq * 16 + r + half * 8;
        AO[((size_t)b * SEQ + m) * DIM + h * DH + l16]      = acc0[r];
        AO[((size_t)b * SEQ + m) * DIM + h * DH + 16 + l16] = acc1[r];
    }
}

// ---------------- tiny final projection 32 -> 3 ----------------
__global__ void head3_kernel(const float* __restrict__ X, const float* __restrict__ Wf3,
                             const float* __restrict__ bf3, float* __restrict__ out) {
    int r = blockIdx.x * blockDim.x + threadIdx.x;
    if (r >= ROWS) return;
    float a0 = bf3[0], a1 = bf3[1], a2 = bf3[2];
    #pragma unroll
    for (int k = 0; k < 32; ++k) {
        float v = X[(size_t)r * 32 + k];
        a0 += v * Wf3[k * 3 + 0];
        a1 += v * Wf3[k * 3 + 1];
        a2 += v * Wf3[k * 3 + 2];
    }
    out[r * 3 + 0] = a0; out[r * 3 + 1] = a1; out[r * 3 + 2] = a2;
}

// ================================================================

static inline void launch_gemm(hipStream_t s, const float* A, const _Float16* Wp,
                               const float* bias, const float* res, float* C,
                               int M, int K, int N, int relu) {
    int tilesN = N / 16;
    if (tilesN % 4 == 0) {
        int groups = (M / 16) * (tilesN / 4);
        int blocks = (groups + 7) / 8;
        if (bias && relu)
            gemm_kernel<4, true,  true,  false><<<blocks, 256, 0, s>>>(A, Wp, bias, res, C, M, K, N);
        else if (bias && res)
            gemm_kernel<4, true,  false, true ><<<blocks, 256, 0, s>>>(A, Wp, bias, res, C, M, K, N);
        else if (bias)
            gemm_kernel<4, true,  false, false><<<blocks, 256, 0, s>>>(A, Wp, bias, res, C, M, K, N);
        else if (res)
            gemm_kernel<4, false, false, true ><<<blocks, 256, 0, s>>>(A, Wp, bias, res, C, M, K, N);
        else
            gemm_kernel<4, false, false, false><<<blocks, 256, 0, s>>>(A, Wp, bias, res, C, M, K, N);
    } else {
        int groups = (M / 16) * (tilesN / 2);
        int blocks = (groups + 7) / 8;
        gemm_kernel<2, true, false, false><<<blocks, 256, 0, s>>>(A, Wp, bias, res, C, M, K, N);
    }
}

extern "C" void kernel_launch(void* const* d_in, const int* in_sizes, int n_in,
                              void* d_out, int out_size, void* d_ws, size_t ws_size,
                              hipStream_t stream) {
    (void)in_sizes; (void)n_in; (void)out_size; (void)ws_size;

    const float* enc         = (const float*)d_in[0];
    const float* point_ratio = (const float*)d_in[4];
    const float* full        = (const float*)d_in[6];
    const unsigned char* msk = (const unsigned char*)d_in[7];
    const float* pos_table   = (const float*)d_in[8];
    const float* label_table = (const float*)d_in[9];
    const float* Wq_s = (const float*)d_in[10];
    const float* Wk_s = (const float*)d_in[11];
    const float* Wv_s = (const float*)d_in[12];
    const float* Wo_s = (const float*)d_in[13];
    const float* Wq_c = (const float*)d_in[14];
    const float* Wk_c = (const float*)d_in[15];
    const float* Wv_c = (const float*)d_in[16];
    const float* Wo_c = (const float*)d_in[17];
    const float* W1   = (const float*)d_in[18];
    const float* b1   = (const float*)d_in[19];
    const float* W2   = (const float*)d_in[20];
    const float* b2   = (const float*)d_in[21];
    const float* Wf1  = (const float*)d_in[22];
    const float* bf1  = (const float*)d_in[23];
    const float* Wf2  = (const float*)d_in[24];
    const float* bf2  = (const float*)d_in[25];
    const float* Wf3  = (const float*)d_in[26];
    const float* bf3  = (const float*)d_in[27];

    // ---- workspace carve-up
    char* w = (char*)d_ws;
    auto allocF = [&](size_t n) { float* p = (float*)w; w += n * sizeof(float); return p; };
    auto allocI = [&](size_t n) { int* p = (int*)w; w += n * sizeof(int); return p; };
    auto allocH = [&](size_t n) { _Float16* p = (_Float16*)w; w += n * sizeof(_Float16); return p; };

    const size_t XSZ = (size_t)ROWS * DIM;
    float* x    = allocF(XSZ);
    float* xn   = allocF(XSZ);
    float* qb   = allocF(XSZ);
    float* kb   = allocF(XSZ);
    float* vb   = allocF(XSZ);
    float* ao   = allocF(XSZ);
    float* memc = allocF(XSZ);
    float* hb   = allocF((size_t)ROWS * FFD);     // FFN hidden / per-batch scores (aliased)
    float* sc   = hb;
    int* labels   = allocI(ROWS);
    int* position = allocI(ROWS);
    _Float16* vt     = allocH((size_t)BS * DIM * SEQ);    // transposed f16 V
    _Float16* wq_s_h = allocH((size_t)NL * DIM * DIM);
    _Float16* wk_s_h = allocH((size_t)NL * DIM * DIM);
    _Float16* wv_s_h = allocH((size_t)NL * DIM * DIM);
    _Float16* wo_s_h = allocH((size_t)NL * DIM * DIM);
    _Float16* wq_c_h = allocH((size_t)NL * DIM * DIM);
    _Float16* wk_c_h = allocH((size_t)NL * DIM * DIM);
    _Float16* wv_c_h = allocH((size_t)NL * DIM * DIM);
    _Float16* wo_c_h = allocH((size_t)NL * DIM * DIM);
    _Float16* w1_h   = allocH((size_t)NL * DIM * FFD);
    _Float16* w2_h   = allocH((size_t)NL * FFD * DIM);
    _Float16* wf1_h  = allocH((size_t)DIM * 128);
    _Float16* wf2_h  = allocH((size_t)128 * 32);
    float* t1  = qb;   // [8192,128]
    float* t1n = kb;
    float* t2  = vb;   // [8192,32]
    float* t2n = ao;

    // ---- pre-pack all GEMM weights into WMMA fragment layout (f16)
    auto packw = [&](const float* src, _Float16* dst, int K, int N, int layers) {
        for (int l = 0; l < layers; ++l) {
            int n = K * N;
            pack_w_kernel<<<(n + 255) / 256, 256, 0, stream>>>(src + (size_t)l * n,
                                                               dst + (size_t)l * n, K, N);
        }
    };
    packw(Wq_s, wq_s_h, DIM, DIM, NL); packw(Wk_s, wk_s_h, DIM, DIM, NL);
    packw(Wv_s, wv_s_h, DIM, DIM, NL); packw(Wo_s, wo_s_h, DIM, DIM, NL);
    packw(Wq_c, wq_c_h, DIM, DIM, NL); packw(Wk_c, wk_c_h, DIM, DIM, NL);
    packw(Wv_c, wv_c_h, DIM, DIM, NL); packw(Wo_c, wo_c_h, DIM, DIM, NL);
    packw(W1, w1_h, DIM, FFD, NL);     packw(W2, w2_h, FFD, DIM, NL);
    packw(Wf1, wf1_h, DIM, 128, 1);    packw(Wf2, wf2_h, 128, 32, 1);

    // ---- pack + embedding + memory slice
    int* lab_out = (int*)d_out + ROWS * 3;
    pack_kernel<<<BS, 256, 0, stream>>>(point_ratio, labels, position, lab_out);
    embed_kernel<<<ROWS, 256, 0, stream>>>(pos_table, label_table, enc, labels, position, x);
    copymem_kernel<<<ROWS, 256, 0, stream>>>(full, memc);

    const float scale = 1.0f / sqrtf((float)DH);

    for (int l = 0; l < NL; ++l) {
        const size_t wo = (size_t)l * DIM * DIM;
        const size_t fo = (size_t)l * DIM * FFD;

        // ===== self attention =====
        ln_kernel<<<ROWS, 256, 0, stream>>>(x, xn, DIM, 0);
        launch_gemm(stream, xn, wq_s_h + wo, nullptr, nullptr, qb, ROWS, DIM, DIM, 0);
        launch_gemm(stream, xn, wk_s_h + wo, nullptr, nullptr, kb, ROWS, DIM, DIM, 0);
        launch_gemm(stream, xn, wv_s_h + wo, nullptr, nullptr, vb, ROWS, DIM, DIM, 0);
        vtrans_kernel<<<ROWS, 256, 0, stream>>>(vb, vt);
        for (int b = 0; b < BS; ++b) {
            score_kernel<<<dim3(128, NH), 256, 0, stream>>>(qb, kb, sc, b, scale);
            softmax_kernel<<<NH * SEQ, 256, 0, stream>>>(sc, labels, nullptr, b);
            av_kernel<<<64, 256, 0, stream>>>(sc, vt, ao, b);
        }
        launch_gemm(stream, ao, wo_s_h + wo, nullptr, x, x, ROWS, DIM, DIM, 0);

        // ===== cross attention =====
        ln_kernel<<<ROWS, 256, 0, stream>>>(x, xn, DIM, 0);
        launch_gemm(stream, xn,   wq_c_h + wo, nullptr, nullptr, qb, ROWS, DIM, DIM, 0);
        launch_gemm(stream, memc, wk_c_h + wo, nullptr, nullptr, kb, ROWS, DIM, DIM, 0);
        launch_gemm(stream, memc, wv_c_h + wo, nullptr, nullptr, vb, ROWS, DIM, DIM, 0);
        vtrans_kernel<<<ROWS, 256, 0, stream>>>(vb, vt);
        for (int b = 0; b < BS; ++b) {
            score_kernel<<<dim3(128, NH), 256, 0, stream>>>(qb, kb, sc, b, scale);
            softmax_kernel<<<NH * SEQ, 256, 0, stream>>>(sc, nullptr, msk, b);
            av_kernel<<<64, 256, 0, stream>>>(sc, vt, ao, b);
        }
        launch_gemm(stream, ao, wo_c_h + wo, nullptr, x, x, ROWS, DIM, DIM, 0);

        // ===== FFN =====
        ln_kernel<<<ROWS, 256, 0, stream>>>(x, xn, DIM, 0);
        launch_gemm(stream, xn, w1_h + fo, b1 + (size_t)l * FFD, nullptr, hb, ROWS, DIM, FFD, 1);
        launch_gemm(stream, hb, w2_h + fo, b2 + (size_t)l * DIM, x, x, ROWS, FFD, DIM, 0);
    }

    // ===== head =====
    launch_gemm(stream, x, wf1_h, bf1, nullptr, t1, ROWS, DIM, 128, 0);
    ln_kernel<<<ROWS, 256, 0, stream>>>(t1, t1n, 128, 1);
    launch_gemm(stream, t1n, wf2_h, bf2, nullptr, t2, ROWS, 128, 32, 0);
    ln_kernel<<<ROWS, 256, 0, stream>>>(t2, t2n, 32, 1);
    head3_kernel<<<(ROWS + 255) / 256, 256, 0, stream>>>(t2n, Wf3, bf3, (float*)d_out);
}
